// Discriminator_66228395704727
// MI455X (gfx1250) — matrix operand order repaired and checked by reading, not compiled, not run on hardware
//
#include <hip/hip_runtime.h>
#include <hip/hip_bf16.h>

typedef __attribute__((ext_vector_type(16))) __bf16 v16bf;
typedef __attribute__((ext_vector_type(8)))  float  v8f;

#define B_   16
#define CIN  8
#define NPOS 4096
#define F_   64
#define FQ   8
#define F2   128
#define KTOT (F2 * NPOS)   /* 524288 */

// workspace layout (byte offsets)
#define OFF_H    (0u)            // h / post-attn h : B*F*N f32 = 16 MiB
#define OFF_Q    (16u << 20)     // q : B*8*N f32 = 2 MiB
#define OFF_K    (18u << 20)     // k : B*8*N f32 = 2 MiB
#define OFF_VBF  (20u << 20)     // v : B*F*N bf16 = 8 MiB
#define OFF_H2T  (28u << 20)     // h2 transposed [c*N][b] f32 = 32 MiB
#define OFF_POOL (60u << 20)     // pooled partial sums: B*F2 f32

__device__ __forceinline__ float leaky(float a) { return (a >= 0.f) ? a : 0.01f * a; }

// ---------------- K0: zero the pooled accumulator ----------------
__global__ void k0_init(float* __restrict__ pooled) {
    int i = blockIdx.x * 256 + threadIdx.x;
    if (i < B_ * F2) pooled[i] = 0.f;
}

// ---------------- K1: fused conv1(1x1)+LeakyReLU and Q/K/V projections ------
__global__ __launch_bounds__(256) void k1_conv1_qkv(
    const float* __restrict__ x,
    const float* __restrict__ w1, const float* __restrict__ b1,
    const float* __restrict__ wq, const float* __restrict__ bq,
    const float* __restrict__ wk, const float* __restrict__ bk,
    const float* __restrict__ wv, const float* __restrict__ bv,
    float* __restrict__ h, float* __restrict__ q, float* __restrict__ k,
    __bf16* __restrict__ vbf)
{
    __shared__ float w1s[F_ * CIN], wqs[FQ * F_], wks[FQ * F_], wvs[F_ * F_];
    __shared__ float b1s[F_], bqs[FQ], bks[FQ], bvs[F_];
    const int tid = threadIdx.x;
    for (int i = tid; i < F_ * CIN; i += 256) w1s[i] = w1[i];
    for (int i = tid; i < FQ * F_; i += 256) { wqs[i] = wq[i]; wks[i] = wk[i]; }
    for (int i = tid; i < F_ * F_; i += 256) wvs[i] = wv[i];
    if (tid < F_) { b1s[tid] = b1[tid]; bvs[tid] = bv[tid]; }
    if (tid < FQ) { bqs[tid] = bq[tid]; bks[tid] = bk[tid]; }
    __syncthreads();

    const int p = blockIdx.x * 256 + tid;           // 0 .. B*N-1
    const int b = p >> 12, n = p & (NPOS - 1);
    const float* xb = x + ((size_t)b * CIN) * NPOS + n;
    float xc[CIN];
#pragma unroll
    for (int c = 0; c < CIN; c++) xc[c] = xb[(size_t)c * NPOS];

    float hv[F_];
#pragma unroll
    for (int o = 0; o < F_; o++) {
        float a = b1s[o];
#pragma unroll
        for (int c = 0; c < CIN; c++) a = fmaf(w1s[o * CIN + c], xc[c], a);
        hv[o] = leaky(a);
    }
    const size_t hb = ((size_t)b * F_) * NPOS + n;
#pragma unroll
    for (int o = 0; o < F_; o++) h[hb + (size_t)o * NPOS] = hv[o];

    const size_t qb = ((size_t)b * FQ) * NPOS + n;
#pragma unroll
    for (int o = 0; o < FQ; o++) {
        float aq = bqs[o], ak = bks[o];
#pragma unroll
        for (int c = 0; c < F_; c++) {
            aq = fmaf(wqs[o * F_ + c], hv[c], aq);
            ak = fmaf(wks[o * F_ + c], hv[c], ak);
        }
        q[qb + (size_t)o * NPOS] = aq;
        k[qb + (size_t)o * NPOS] = ak;
    }
#pragma unroll 4
    for (int o = 0; o < F_; o++) {
        float a = bvs[o];
#pragma unroll
        for (int c = 0; c < F_; c++) a = fmaf(wvs[o * F_ + c], hv[c], a);
        vbf[hb + (size_t)o * NPOS] = (__bf16)a;
    }
}

// ---------------- K2: flash attention, P*V on v_wmma_f32_16x16x32_bf16 ------
// One wave32 per (batch, 16-query tile). Oᵀ = P(16x32) x Vᵀ(32x16), 4 channel blocks.
__global__ __launch_bounds__(32) void k2_attn(
    const float* __restrict__ q, const float* __restrict__ k,
    const __bf16* __restrict__ vbf, float* __restrict__ h,
    const float* __restrict__ gamma)
{
    __shared__ __align__(16) float kt[CIN][32];
    __shared__ float smax[16], sden[16];
    const int b    = blockIdx.x;
    const int n0   = blockIdx.y * 16;
    const int lane = threadIdx.x;
    const int row  = lane & 15;     // query row this lane owns in A / stats
    const int half = lane >> 4;

    const float* qb = q + (size_t)b * FQ * NPOS;
    const float* kb = k + (size_t)b * FQ * NPOS;
    float qr[CIN];
#pragma unroll
    for (int c = 0; c < CIN; c++) qr[c] = qb[(size_t)c * NPOS + n0 + row];

    // ---- pass 1: exact row max over all 4096 keys ----
    float rmax = -3.0e38f;
    for (int m0 = 0; m0 < NPOS; m0 += 32) {
        { // stage k tile 8x32 into LDS (each lane: 8 consecutive floats)
            const int c = lane & 7, g = lane >> 3;
            const float* src = kb + (size_t)c * NPOS + m0 + g * 8;
            float4 a = *(const float4*)src, bb = *(const float4*)(src + 4);
            *(float4*)&kt[c][g * 8] = a; *(float4*)&kt[c][g * 8 + 4] = bb;
        }
        __syncthreads();
#pragma unroll
        for (int e = 0; e < 16; e++) {
            const int off = half * 8 + (e < 8 ? e : e + 8);   // WMMA A-layout key index
            float s = 0.f;
#pragma unroll
            for (int c = 0; c < CIN; c++) s = fmaf(qr[c], kt[c][off], s);
            rmax = fmaxf(rmax, s);
        }
        __syncthreads();
    }
    rmax = fmaxf(rmax, __shfl_xor(rmax, 16));
    if (lane < 16) smax[lane] = rmax;
    __syncthreads();
    const float rmx = smax[row];

    // ---- pass 2: unnormalized exp, accumulate Oᵀ with WMMA, track denominator
    v8f acc0 = {}, acc1 = {}, acc2 = {}, acc3 = {};
    float pden = 0.f;
    const __bf16* vb = vbf + (size_t)b * F_ * NPOS;
    const int c0 = row;   // channel within block for B operand / D column
    for (int m0 = 0; m0 < NPOS; m0 += 32) {
        {
            const int c = lane & 7, g = lane >> 3;
            const float* src = kb + (size_t)c * NPOS + m0 + g * 8;
            float4 a = *(const float4*)src, bb = *(const float4*)(src + 4);
            *(float4*)&kt[c][g * 8] = a; *(float4*)&kt[c][g * 8 + 4] = bb;
        }
        __syncthreads();
        v16bf A;
#pragma unroll
        for (int e = 0; e < 16; e++) {
            const int off = half * 16 + e;
            float s = 0.f;
#pragma unroll
            for (int c = 0; c < CIN; c++) s = fmaf(qr[c], kt[c][off], s);
            const float pe = __expf(s - rmx);
            pden += pe;
            A[e] = (__bf16)pe;
        }
        // B operand = Vᵀ tile: 16 consecutive bf16 per lane, naturally aligned
        const __bf16* vbase = vb + m0 + half * 16;
        v16bf B0 = *(const v16bf*)(vbase + (size_t)(c0     ) * NPOS);
        v16bf B1 = *(const v16bf*)(vbase + (size_t)(c0 + 16) * NPOS);
        v16bf B2 = *(const v16bf*)(vbase + (size_t)(c0 + 32) * NPOS);
        v16bf B3 = *(const v16bf*)(vbase + (size_t)(c0 + 48) * NPOS);
        acc0 = __builtin_amdgcn_wmma_f32_16x16x32_bf16(false, A, false, B0, (short)0, acc0, false, false);
        acc1 = __builtin_amdgcn_wmma_f32_16x16x32_bf16(false, A, false, B1, (short)0, acc1, false, false);
        acc2 = __builtin_amdgcn_wmma_f32_16x16x32_bf16(false, A, false, B2, (short)0, acc2, false, false);
        acc3 = __builtin_amdgcn_wmma_f32_16x16x32_bf16(false, A, false, B3, (short)0, acc3, false, false);
        __syncthreads();
    }
    pden += __shfl_xor(pden, 16);
    if (lane < 16) sden[lane] = pden;
    __syncthreads();

    // epilogue: h = gamma * (Oᵀ/den) + h ; D row r maps to query r + 8*half
    const float g = gamma[0];
    float* hbp = h + (size_t)b * F_ * NPOS;
#pragma unroll
    for (int r = 0; r < 8; r++) {
        const int qrow = r + half * 8;
        const float inv = 1.0f / sden[qrow];
        const size_t i0 = (size_t)c0 * NPOS + (size_t)(n0 + qrow);
        hbp[i0                   ] = fmaf(g, acc0[r] * inv, hbp[i0]);
        hbp[i0 + (size_t)16*NPOS ] = fmaf(g, acc1[r] * inv, hbp[i0 + (size_t)16*NPOS]);
        hbp[i0 + (size_t)32*NPOS ] = fmaf(g, acc2[r] * inv, hbp[i0 + (size_t)32*NPOS]);
        hbp[i0 + (size_t)48*NPOS ] = fmaf(g, acc3[r] * inv, hbp[i0 + (size_t)48*NPOS]);
    }
}

// ---------------- K3: conv2(1x1)+LeakyReLU, store h2 transposed [o*N+n][b] --
__global__ __launch_bounds__(256) void k3_conv2(
    const float* __restrict__ h, const float* __restrict__ w2,
    const float* __restrict__ b2, float* __restrict__ h2t)
{
    __shared__ float w2s[F2 * F_];
    __shared__ float b2s[F2];
    const int tid = threadIdx.x;
    for (int i = tid; i < F2 * F_; i += 256) w2s[i] = w2[i];
    if (tid < F2) b2s[tid] = b2[tid];
    __syncthreads();

    const int p = blockIdx.x * 256 + tid;
    const int b = p & 15, n = p >> 4;     // b fastest -> coalesced h2t stores
    const float* hb = h + (size_t)b * F_ * NPOS + n;
    float hc[F_];
#pragma unroll
    for (int c = 0; c < F_; c++) hc[c] = hb[(size_t)c * NPOS];
    for (int o = 0; o < F2; o++) {
        float a = b2s[o];
#pragma unroll
        for (int c = 0; c < F_; c++) a = fmaf(w2s[o * F_ + c], hc[c], a);
        h2t[((size_t)o * NPOS + n) * B_ + b] = leaky(a);
    }
}

// ---------------- K4: pool conv — stream wp once (268 MB), h2t from L2 ------
__global__ __launch_bounds__(256) void k4_pool(
    const float* __restrict__ h2t, const float* __restrict__ wp,
    float* __restrict__ pooled)
{
    __shared__ float red[B_];
    const int tid = threadIdx.x;
    const int o = blockIdx.x;       // pool output channel
    const int chunk = blockIdx.y;   // K chunk of 4096 (c,n) elements
    if (tid < B_) red[tid] = 0.f;
    __syncthreads();
    float acc[B_];
#pragma unroll
    for (int b = 0; b < B_; b++) acc[b] = 0.f;

    const float* wpo = wp + (size_t)o * KTOT + (size_t)chunk * 4096;
    const float* h2b = h2t + (size_t)chunk * 4096 * B_;
    for (int j = 0; j < 16; j++) {
        const int e = j * 256 + tid;
        const float w = wpo[e];
        __builtin_prefetch(wpo + e + 2048, 0, 0);      // global_prefetch_b8
        const float4* hv4 = (const float4*)(h2b + (size_t)e * B_);
#pragma unroll
        for (int t = 0; t < 4; t++) {
            const float4 hv = hv4[t];
            acc[t * 4 + 0] = fmaf(w, hv.x, acc[t * 4 + 0]);
            acc[t * 4 + 1] = fmaf(w, hv.y, acc[t * 4 + 1]);
            acc[t * 4 + 2] = fmaf(w, hv.z, acc[t * 4 + 2]);
            acc[t * 4 + 3] = fmaf(w, hv.w, acc[t * 4 + 3]);
        }
    }
#pragma unroll
    for (int b = 0; b < B_; b++) atomicAdd(&red[b], acc[b]);
    __syncthreads();
    if (tid < B_) atomicAdd(&pooled[(size_t)tid * F2 + o], red[tid]);
}

// ---------------- K5: final linear (B,128) x (128,) + biases ----------------
__global__ __launch_bounds__(32) void k5_out(
    const float* __restrict__ pooled, const float* __restrict__ bp,
    const float* __restrict__ wo, const float* __restrict__ bo,
    float* __restrict__ out)
{
    const int b = threadIdx.x;
    if (b < B_) {
        float a = bo[0];
        for (int o = 0; o < F2; o++) a = fmaf(pooled[b * F2 + o] + bp[o], wo[o], a);
        out[b] = a;
    }
}

extern "C" void kernel_launch(void* const* d_in, const int* in_sizes, int n_in,
                              void* d_out, int out_size, void* d_ws, size_t ws_size,
                              hipStream_t stream)
{
    (void)in_sizes; (void)n_in; (void)out_size; (void)ws_size;
    const float* x  = (const float*)d_in[0];
    const float* w1 = (const float*)d_in[1];
    const float* b1 = (const float*)d_in[2];
    const float* wq = (const float*)d_in[3];
    const float* bq = (const float*)d_in[4];
    const float* wk = (const float*)d_in[5];
    const float* bk = (const float*)d_in[6];
    const float* wv = (const float*)d_in[7];
    const float* bv = (const float*)d_in[8];
    const float* gamma = (const float*)d_in[9];
    const float* w2 = (const float*)d_in[10];
    const float* b2 = (const float*)d_in[11];
    const float* wp = (const float*)d_in[12];
    const float* bp = (const float*)d_in[13];
    const float* wo = (const float*)d_in[14];
    const float* bo = (const float*)d_in[15];

    char* ws = (char*)d_ws;
    float*  h      = (float*)(ws + OFF_H);
    float*  q      = (float*)(ws + OFF_Q);
    float*  kk     = (float*)(ws + OFF_K);
    __bf16* vbf    = (__bf16*)(ws + OFF_VBF);
    float*  h2t    = (float*)(ws + OFF_H2T);
    float*  pooled = (float*)(ws + OFF_POOL);

    k0_init<<<dim3((B_ * F2 + 255) / 256), 256, 0, stream>>>(pooled);
    k1_conv1_qkv<<<dim3((B_ * NPOS) / 256), 256, 0, stream>>>(
        x, w1, b1, wq, bq, wk, bk, wv, bv, h, q, kk, vbf);
    k2_attn<<<dim3(B_, NPOS / 16), 32, 0, stream>>>(q, kk, vbf, h, gamma);
    k3_conv2<<<dim3((B_ * NPOS) / 256), 256, 0, stream>>>(h, w2, b2, h2t);
    k4_pool<<<dim3(F2, 128), 256, 0, stream>>>(h2t, wp, pooled);
    k5_out<<<1, 32, 0, stream>>>(pooled, bp, wo, bo, (float*)d_out);
}
